// MultiHeadMapAttentionV2_69801808494848
// MI455X (gfx1250) — matrix operand order repaired and checked by reading, not compiled, not run on hardware
//
#include <hip/hip_runtime.h>
#include <hip/hip_bf16.h>

#define BB   256
#define CC   1024
#define SS   14
#define HW   196
#define NTOK 197
#define NH   8
#define DK   64

typedef __attribute__((ext_vector_type(8)))  __bf16 v8bf;
typedef __attribute__((ext_vector_type(16))) __bf16 v16bf;
typedef __attribute__((ext_vector_type(8)))  float  v8f;

__device__ __forceinline__ float wave_reduce_sum(float v) {
#pragma unroll
  for (int off = 16; off > 0; off >>= 1) v += __shfl_down(v, off, 32);
  return v;
}

// ---------------------------------------------------------------------------
// Spatial mean of loss_map: (B,8,196) -> (B,8)
__global__ void __launch_bounds__(256) lmean_kernel(const float* __restrict__ lm,
                                                    float* __restrict__ lmean) {
  int b = blockIdx.x;
  int c = threadIdx.x >> 5;          // 8 waves, one channel each
  int lane = threadIdx.x & 31;
  const float* p = lm + ((size_t)b * 8 + c) * HW;
  float s = 0.f;
  for (int i = lane; i < HW; i += 32) s += p[i];
  s = wave_reduce_sum(s);
  if (lane == 0) lmean[b * 8 + c] = s * (1.0f / (float)HW);
}

// Spatial mean of feature_map: (B,1024,196) -> (B,1024)
__global__ void __launch_bounds__(256) fmean_kernel(const float* __restrict__ fm,
                                                    float* __restrict__ fmean) {
  int b = blockIdx.x;
  int c = blockIdx.y * 8 + (threadIdx.x >> 5);
  int lane = threadIdx.x & 31;
  const float* p = fm + ((size_t)b * CC + c) * HW;
  float s = 0.f;
  for (int i = lane; i < HW; i += 32) s += p[i];
  s = wave_reduce_sum(s);
  if (lane == 0) fmean[(size_t)b * CC + c] = s * (1.0f / (float)HW);
}

// ---------------------------------------------------------------------------
// dense helper for the collapsed conv-MLP (mean(conv(x)) == conv(mean(x)))
__device__ __forceinline__ void dense_layer(const float* __restrict__ w,
                                            const float* __restrict__ bias,
                                            const float* __restrict__ in,
                                            float* __restrict__ out,
                                            int cin, int cout, int t, int nt) {
  for (int o = t; o < cout; o += nt) {
    const float* wr = w + (size_t)o * cin;
    float s = bias[o];
    for (int i = 0; i < cin; ++i) s += wr[i] * in[i];
    out[o] = s;
  }
}

// Per batch row: 8->32->64->128->256->1024 MLP, + pos_q[0] -> queries
__global__ void __launch_bounds__(256) mlp_queries_kernel(
    const float* __restrict__ lmean,
    const float* __restrict__ w1, const float* __restrict__ b1,
    const float* __restrict__ w2, const float* __restrict__ b2,
    const float* __restrict__ w3, const float* __restrict__ b3,
    const float* __restrict__ w4, const float* __restrict__ b4,
    const float* __restrict__ w5, const float* __restrict__ b5,
    const float* __restrict__ pos_q,
    float* __restrict__ q_f32, __bf16* __restrict__ q_bf16) {
  __shared__ float hA[1024];
  __shared__ float hB[1024];
  int b = blockIdx.x, t = threadIdx.x;
  if (t < 8) hA[t] = lmean[b * 8 + t];
  __syncthreads();
  dense_layer(w1, b1, hA, hB, 8, 32, t, 256);   __syncthreads();
  dense_layer(w2, b2, hB, hA, 32, 64, t, 256);  __syncthreads();
  dense_layer(w3, b3, hA, hB, 64, 128, t, 256); __syncthreads();
  dense_layer(w4, b4, hB, hA, 128, 256, t, 256);__syncthreads();
  dense_layer(w5, b5, hA, hB, 256, 1024, t, 256);__syncthreads();
#pragma unroll
  for (int i = 0; i < 4; ++i) {
    int c = t + i * 256;
    float v = hB[c] + pos_q[c];           // pos_q row 0
    q_f32[(size_t)b * CC + c] = v;
    q_bf16[(size_t)b * CC + c] = (__bf16)v;
  }
}

// ---------------------------------------------------------------------------
// X token 0 = feature mean + pos_kv[0]
__global__ void __launch_bounds__(256) token0_kernel(const float* __restrict__ fmean,
                                                     const float* __restrict__ pos_kv,
                                                     __bf16* __restrict__ X) {
  int b = blockIdx.x;
  for (int c = threadIdx.x; c < CC; c += 256)
    X[(size_t)b * NTOK * CC + c] = (__bf16)(fmean[(size_t)b * CC + c] + pos_kv[c]);
}

// Transpose (C,HW)->(HW,C) per batch, add pos_kv, emit bf16 tokens 1..196
__global__ void __launch_bounds__(256) build_x_kernel(const float* __restrict__ fm,
                                                      const float* __restrict__ pos_kv,
                                                      __bf16* __restrict__ X) {
  __shared__ float tile[32][33];
  int b = blockIdx.x;
  int c0 = blockIdx.y * 32;
  int hw0 = blockIdx.z * 32;
  int tx = threadIdx.x & 31, ty = threadIdx.x >> 5;   // 32 x 8
#pragma unroll
  for (int i = 0; i < 4; ++i) {
    int c = c0 + ty + i * 8;
    int hw = hw0 + tx;
    float v = 0.f;
    if (hw < HW) v = fm[((size_t)b * CC + c) * HW + hw];
    tile[ty + i * 8][tx] = v;
  }
  __syncthreads();
#pragma unroll
  for (int i = 0; i < 4; ++i) {
    int hw = hw0 + ty + i * 8;
    int c = c0 + tx;
    if (hw < HW)
      X[((size_t)b * NTOK + 1 + hw) * CC + c] =
          (__bf16)(tile[tx][ty + i * 8] + pos_kv[(size_t)(1 + hw) * CC + c]);
  }
}

// f32 -> bf16 weight conversion
__global__ void f32_to_bf16_kernel(const float* __restrict__ src,
                                   __bf16* __restrict__ dst, int n) {
  int i = blockIdx.x * blockDim.x + threadIdx.x;
  if (i < n) dst[i] = (__bf16)src[i];
}

// ---------------------------------------------------------------------------
// Wave32 WMMA GEMM, 64x128 block tile, 4 M-subtiles per wave (B-frag reuse x4).
// Cout(M,N) f32 = A(M,K) bf16 * W(N,K)^T bf16 + bias
// grid = (M/64, N/128), block = 256 (8 waves). A staged in 32KB LDS chunks
// (64 rows x 256 k) so no >64KB dynamic LDS is needed. K % 256 == 0.
// Per k-step: issue ALL ds_loads first, then 4 independent WMMAs back-to-back
// so the compiler can use partial dscnt waits and keep the XDL pipe fed.
#define GEMM_KC 256
#define GEMM_MT 4
__global__ void __launch_bounds__(256) wmma_gemm_bf16_kernel(
    const __bf16* __restrict__ A, const __bf16* __restrict__ W,
    const float* __restrict__ bias, float* __restrict__ Cout,
    int M, int N, int K) {
  __shared__ __bf16 lds_a[64 * GEMM_KC];            // 32 KB
  const int tid  = threadIdx.x;
  const int wave = tid >> 5;
  const int lane = tid & 31;
  const int row0 = blockIdx.x * 64;
  const int col0 = blockIdx.y * 128 + wave * 16;

  const int m  = lane & 15;   // A row within subtile / D row-group
  const int hi = lane >> 4;   // half select per 16-bit WMMA layout
  const int n  = lane & 15;   // B/D column

  v8f acc[GEMM_MT];
  float bval = bias[col0 + n];
#pragma unroll
  for (int s = 0; s < GEMM_MT; ++s)
#pragma unroll
    for (int r = 0; r < 8; ++r) acc[s][r] = bval;

  const __bf16* wrow = W + (size_t)(col0 + n) * K;

  for (int kb = 0; kb < K; kb += GEMM_KC) {
    // stage 64 x 256 A chunk into LDS (128-bit chunks, coalesced)
    for (int i = tid; i < 64 * (GEMM_KC / 8); i += 256) {
      int r  = i >> 5;                 // / (GEMM_KC/8)
      int kc = (i & 31) << 3;
      *reinterpret_cast<v8bf*>(&lds_a[r * GEMM_KC + kc]) =
          *reinterpret_cast<const v8bf*>(&A[(size_t)(row0 + r) * K + kb + kc]);
    }
    __syncthreads();

#pragma unroll 2
    for (int k0 = 0; k0 < GEMM_KC; k0 += 32) {
      // B 32x16 fragment: lane<16 -> k=k0..k0+15, lane>=16 -> k+16..k+31, col n
      v16bf bfrag = *reinterpret_cast<const v16bf*>(&wrow[kb + k0 + hi * 16]);

      // A 16x32 fragments: lanes 0-15 row m k=k0..k0+15;
      // lanes 16-31 same rows, k+16..k+31. Load all 4 subtiles first.
      v16bf afrag[GEMM_MT];
#pragma unroll
      for (int s = 0; s < GEMM_MT; ++s) {
        const __bf16* ar = &lds_a[(s * 16 + m) * GEMM_KC + k0 + hi * 16];
        v8bf alo = *reinterpret_cast<const v8bf*>(ar);
        v8bf ahi = *reinterpret_cast<const v8bf*>(ar + 8);
        afrag[s] = __builtin_shufflevector(alo, ahi,
            0,1,2,3,4,5,6,7,8,9,10,11,12,13,14,15);
      }
      // 4 independent WMMAs (distinct accumulators) reuse one B fragment
#pragma unroll
      for (int s = 0; s < GEMM_MT; ++s)
        acc[s] = __builtin_amdgcn_wmma_f32_16x16x32_bf16(
            false, afrag[s], false, bfrag, (short)0, acc[s], false, false);
    }
    __syncthreads();
  }

  // D 16x16 f32: VGPR r -> row (s*16 + r + 8*hi), col n. Output is consumed
  // exactly once downstream and KV (206MB) exceeds L2 -> non-temporal stores.
#pragma unroll
  for (int s = 0; s < GEMM_MT; ++s)
#pragma unroll
    for (int r = 0; r < 8; ++r) {
      int mm = s * 16 + r + hi * 8;
      __builtin_nontemporal_store(
          acc[s][r], &Cout[(size_t)(row0 + mm) * N + col0 + n]);
    }
}

// ---------------------------------------------------------------------------
// Attention, q-length 1. KV layout: (B*197, 1024) f32; cols 0-511 = K, 512+ = V
__global__ void __launch_bounds__(256) attention_kernel(
    const float* __restrict__ q,       // (B, 512) f32
    const float* __restrict__ KV,
    __bf16* __restrict__ attout) {     // (B, 512) bf16
  __shared__ float logits[NTOK];
  __shared__ float qs[64];
  __shared__ float red[8];
  __shared__ float red2[8];
  __shared__ float outacc[4][64];
  int b = blockIdx.x, h = blockIdx.y;
  int t = threadIdx.x, lane = t & 31, wave = t >> 5;

  if (t < 64) qs[t] = q[(size_t)b * 512 + h * 64 + t];
  __syncthreads();

  const float* Kbase = KV + (size_t)b * NTOK * CC + h * 64;
  for (int nn = wave; nn < NTOK; nn += 8) {
    const float* kr = Kbase + (size_t)nn * CC;
    float s = __builtin_nontemporal_load(&kr[lane]) * qs[lane] +
              __builtin_nontemporal_load(&kr[lane + 32]) * qs[lane + 32];
    s = wave_reduce_sum(s);
    if (lane == 0) logits[nn] = s * 0.125f;     // 1/sqrt(64)
  }
  __syncthreads();

  // softmax: block max
  float m = -1e30f;
  if (t < NTOK) m = logits[t];
#pragma unroll
  for (int off = 16; off > 0; off >>= 1) m = fmaxf(m, __shfl_down(m, off, 32));
  if (lane == 0) red[wave] = m;
  __syncthreads();
  if (t == 0) {
    float mm = red[0];
    for (int w = 1; w < 8; ++w) mm = fmaxf(mm, red[w]);
    red[0] = mm;
  }
  __syncthreads();
  m = red[0];

  float s = 0.f;
  if (t < NTOK) { float e = __expf(logits[t] - m); logits[t] = e; s = e; }
  __syncthreads();
#pragma unroll
  for (int off = 16; off > 0; off >>= 1) s += __shfl_down(s, off, 32);
  if (lane == 0) red2[wave] = s;
  __syncthreads();
  if (t == 0) {
    float ss = 0.f;
    for (int w = 0; w < 8; ++w) ss += red2[w];
    red2[0] = ss;
  }
  __syncthreads();
  float inv = 1.0f / red2[0];

  // out[d] = sum_n logits[n] * V[n][d]
  int d = t & 63, g = t >> 6;                    // 4 groups of 64
  const float* Vbase = KV + (size_t)b * NTOK * CC + 512 + h * 64;
  float acc = 0.f;
  for (int nn = g; nn < NTOK; nn += 4)
    acc += logits[nn] * __builtin_nontemporal_load(&Vbase[(size_t)nn * CC + d]);
  outacc[g][d] = acc;
  __syncthreads();
  if (g == 0) {
    float o = (outacc[0][d] + outacc[1][d] + outacc[2][d] + outacc[3][d]) * inv;
    attout[(size_t)b * 512 + h * 64 + d] = (__bf16)o;
  }
}

// ---------------------------------------------------------------------------
// Residual + LayerNorm over C=1024, one block per batch row
__global__ void __launch_bounds__(256) ln_kernel(
    const float* __restrict__ queries, const float* __restrict__ proj,
    const float* __restrict__ g, const float* __restrict__ beta,
    float* __restrict__ out) {
  __shared__ float red[8];
  int b = blockIdx.x, t = threadIdx.x, lane = t & 31, wave = t >> 5;
  float vals[4];
  float s = 0.f;
#pragma unroll
  for (int i = 0; i < 4; ++i) {
    int c = t + i * 256;
    vals[i] = queries[(size_t)b * CC + c] + proj[(size_t)b * CC + c];
    s += vals[i];
  }
#pragma unroll
  for (int off = 16; off > 0; off >>= 1) s += __shfl_down(s, off, 32);
  if (lane == 0) red[wave] = s;
  __syncthreads();
  if (t == 0) {
    float ss = 0.f;
    for (int w = 0; w < 8; ++w) ss += red[w];
    red[0] = ss * (1.0f / (float)CC);
  }
  __syncthreads();
  float mu = red[0];
  __syncthreads();
  float v = 0.f;
#pragma unroll
  for (int i = 0; i < 4; ++i) { float d = vals[i] - mu; v += d * d; }
#pragma unroll
  for (int off = 16; off > 0; off >>= 1) v += __shfl_down(v, off, 32);
  if (lane == 0) red[wave] = v;
  __syncthreads();
  if (t == 0) {
    float ss = 0.f;
    for (int w = 0; w < 8; ++w) ss += red[w];
    red[0] = ss * (1.0f / (float)CC);
  }
  __syncthreads();
  float rstd = rsqrtf(red[0] + 1e-5f);
#pragma unroll
  for (int i = 0; i < 4; ++i) {
    int c = t + i * 256;
    out[(size_t)b * CC + c] = (vals[i] - mu) * rstd * g[c] + beta[c];
  }
}

// ---------------------------------------------------------------------------
extern "C" void kernel_launch(void* const* d_in, const int* in_sizes, int n_in,
                              void* d_out, int out_size, void* d_ws, size_t ws_size,
                              hipStream_t stream) {
  (void)in_sizes; (void)n_in; (void)out_size; (void)ws_size;
  const float* fm    = (const float*)d_in[0];
  const float* lm    = (const float*)d_in[1];
  const float* w1 = (const float*)d_in[2];  const float* b1 = (const float*)d_in[3];
  const float* w2 = (const float*)d_in[4];  const float* b2 = (const float*)d_in[5];
  const float* w3 = (const float*)d_in[6];  const float* b3 = (const float*)d_in[7];
  const float* w4 = (const float*)d_in[8];  const float* b4 = (const float*)d_in[9];
  const float* w5 = (const float*)d_in[10]; const float* b5 = (const float*)d_in[11];
  const float* wq = (const float*)d_in[12]; const float* bq = (const float*)d_in[13];
  const float* wk = (const float*)d_in[14]; const float* bk = (const float*)d_in[15];
  const float* wv = (const float*)d_in[16]; const float* bv = (const float*)d_in[17];
  const float* wo = (const float*)d_in[18]; const float* bo = (const float*)d_in[19];
  const float* ln_g = (const float*)d_in[20];
  const float* ln_b = (const float*)d_in[21];
  const float* pos_kv = (const float*)d_in[22];
  const float* pos_q  = (const float*)d_in[23];
  float* out = (float*)d_out;

  char* ws = (char*)d_ws;
  size_t off = 0;
  auto alloc = [&](size_t bytes) -> void* {
    void* p = ws + off;
    off += (bytes + 255) & ~(size_t)255;
    return p;
  };

  const size_t Mtok = (size_t)BB * NTOK;                       // 50432 rows
  __bf16* X_bf    = (__bf16*)alloc(Mtok * CC * 2);             // tokens, bf16
  float*  KV_f32  = (float*) alloc(Mtok * CC * 4);             // K|V, f32
  float*  fmean   = (float*) alloc((size_t)BB * CC * 4);
  float*  lmean   = (float*) alloc((size_t)BB * 8 * 4);
  float*  qrs_f32 = (float*) alloc((size_t)BB * CC * 4);       // queries f32
  __bf16* qrs_bf  = (__bf16*)alloc((size_t)BB * CC * 2);       // queries bf16
  __bf16* wkv_bf  = (__bf16*)alloc((size_t)1024 * 1024 * 2);   // [wk;wv] bf16
  __bf16* wq_bf   = (__bf16*)alloc((size_t)512 * 1024 * 2);
  __bf16* wo_bf   = (__bf16*)alloc((size_t)1024 * 512 * 2);
  float*  bias_kv = (float*) alloc(1024 * 4);
  float*  q_proj  = (float*) alloc((size_t)BB * 512 * 4);      // q after wq
  __bf16* att_bf  = (__bf16*)alloc((size_t)BB * 512 * 2);      // attention out
  float*  proj    = (float*) alloc((size_t)BB * CC * 4);       // after wo

  // --- spatial means
  lmean_kernel<<<BB, 256, 0, stream>>>(lm, lmean);
  fmean_kernel<<<dim3(BB, CC / 8), 256, 0, stream>>>(fm, fmean);

  // --- collapsed conv-MLP -> queries
  mlp_queries_kernel<<<BB, 256, 0, stream>>>(lmean, w1, b1, w2, b2, w3, b3,
                                             w4, b4, w5, b5, pos_q,
                                             qrs_f32, qrs_bf);

  // --- build token matrix X (bf16)
  token0_kernel<<<BB, 256, 0, stream>>>(fmean, pos_kv, X_bf);
  build_x_kernel<<<dim3(BB, 32, 7), 256, 0, stream>>>(fm, pos_kv, X_bf);

  // --- weights to bf16 ([wk;wv] concatenated so K+V is one GEMM)
  {
    int n = 512 * 1024, blk = 256, grd = (n + blk - 1) / blk;
    f32_to_bf16_kernel<<<grd, blk, 0, stream>>>(wk, wkv_bf, n);
    f32_to_bf16_kernel<<<grd, blk, 0, stream>>>(wv, wkv_bf + (size_t)512 * 1024, n);
    f32_to_bf16_kernel<<<grd, blk, 0, stream>>>(wq, wq_bf, n);
    f32_to_bf16_kernel<<<grd, blk, 0, stream>>>(wo, wo_bf, 1024 * 512);
  }
  hipMemcpyAsync(bias_kv, bk, 512 * sizeof(float), hipMemcpyDeviceToDevice, stream);
  hipMemcpyAsync(bias_kv + 512, bv, 512 * sizeof(float), hipMemcpyDeviceToDevice, stream);

  // --- q projection: (256,1024) x (512,1024)^T -> (256,512)
  wmma_gemm_bf16_kernel<<<dim3(BB / 64, 512 / 128), 256, 0, stream>>>(
      qrs_bf, wq_bf, bq, q_proj, BB, 512, 1024);

  // --- K|V projection: (50432,1024) x (1024,1024)^T -> (50432,1024) [the big one]
  wmma_gemm_bf16_kernel<<<dim3((int)(Mtok / 64), 1024 / 128), 256, 0, stream>>>(
      X_bf, wkv_bf, bias_kv, KV_f32, (int)Mtok, 1024, 1024);

  // --- attention (q-len 1) per (b,h)
  attention_kernel<<<dim3(BB, NH), 256, 0, stream>>>(q_proj, KV_f32, att_bf);

  // --- output projection: (256,512) x (1024,512)^T -> (256,1024)
  wmma_gemm_bf16_kernel<<<dim3(BB / 64, 1024 / 128), 256, 0, stream>>>(
      att_bf, wo_bf, bo, proj, BB, 1024, 512);

  // --- residual + layernorm
  ln_kernel<<<BB, 256, 0, stream>>>(qrs_f32, proj, ln_g, ln_b, out);
}
